// DeepFM_55439437857626
// MI455X (gfx1250) — matrix operand order repaired and checked by reading, not compiled, not run on hardware
//
#include <hip/hip_runtime.h>
#include <hip/hip_bf16.h>
#include <math.h>

// ---------------- constants ----------------
#define BATCH       16384
#define FACE_K      64
#define VISUAL_DIM  2048
#define HIDDEN      512
#define NUM_USERS   1000000
#define NUM_AGES    8
#define NUM_ATTR    40
#define VOCAB       (NUM_USERS + 24 + 2 + NUM_AGES + NUM_ATTR)
#define X0_LD       (6 * FACE_K)   // 384: 5*64 embeddings + 64 visual proj

typedef __bf16 bf16_t;
typedef __attribute__((ext_vector_type(16))) __bf16 v16bf;
typedef __attribute__((ext_vector_type(8)))  __bf16 v8bf;
typedef __attribute__((ext_vector_type(8)))  float  v8f;

// workspace layout (bytes)
#define OFF_FM  0u                          // float [B,2]            131072
#define OFF_X0  131072u                     // bf16  [B,384]          12582912
#define OFF_X1  (OFF_X0 + 12582912u)        // bf16  [B,512]          16777216
#define OFF_X2  (OFF_X1 + 16777216u)        // bf16  [B,512]          16777216
#define OFF_WV  (OFF_X2 + 16777216u)        // bf16  [64,2048]        262144
#define OFF_W1  (OFF_WV + 262144u)          // bf16  [512,384]        393216
#define OFF_W2  (OFF_W1 + 393216u)          // bf16  [512,512]        524288

// ---------------- fp32 -> bf16 weight conversion ----------------
__global__ __launch_bounds__(256) void cvt_f32_bf16(const float* __restrict__ src,
                                                    bf16_t* __restrict__ dst, int n) {
  for (int i = blockIdx.x * blockDim.x + threadIdx.x; i < n; i += gridDim.x * blockDim.x)
    dst[i] = (bf16_t)src[i];
}

// ---------------- embedding gather + FM linear + second order ----------------
// one wave per batch row; 8 waves / block
__global__ __launch_bounds__(256) void embed_fm_kernel(
    const int* __restrict__ user_id, const int* __restrict__ hour,
    const int* __restrict__ gender, const int* __restrict__ age,
    const int* __restrict__ attribute, const float* __restrict__ scale,
    const float* __restrict__ user_emb, const float* __restrict__ hour_emb,
    const float* __restrict__ gender_emb, const float* __restrict__ age_emb,
    const float* __restrict__ attr_emb, const float* __restrict__ fm_w,
    const float* __restrict__ fm_b, bf16_t* __restrict__ x0,
    float* __restrict__ fm) {
  const int wv   = threadIdx.x >> 5;
  const int lane = threadIdx.x & 31;
  const int b    = blockIdx.x * 8 + wv;
  if (b >= BATCH) return;

  const int uid = user_id[b];
  const int hr  = hour[b];
  const int gd  = gender[b];
  const int ag  = age[b];
  const int at  = attribute[b];
  const float sc = scale[b];

  bf16_t* xr = x0 + (size_t)b * X0_LD;
  float so = 0.0f;
#pragma unroll
  for (int t = 0; t < 2; ++t) {
    const int k = lane + t * 32;
    const float u  = user_emb[(size_t)uid * FACE_K + k];
    const float h  = hour_emb[hr * FACE_K + k];
    const float fg = sc * tanhf(gender_emb[gd * FACE_K + k]);
    const float fa = sc * tanhf(attr_emb[at * FACE_K + k]);
    const float fe = sc * tanhf(age_emb[ag * FACE_K + k]);
    // concat order: user, hour, gender, attr, age
    xr[0 * FACE_K + k] = (bf16_t)u;
    xr[1 * FACE_K + k] = (bf16_t)h;
    xr[2 * FACE_K + k] = (bf16_t)fg;
    xr[3 * FACE_K + k] = (bf16_t)fa;
    xr[4 * FACE_K + k] = (bf16_t)fe;
    const float s  = u + h + fg + fa + fe;
    const float sq = u * u + h * h + fg * fg + fa * fa + fe * fe;
    so += s * s - sq;
  }
#pragma unroll
  for (int off = 16; off > 0; off >>= 1) so += __shfl_xor(so, off, 32);

  if (lane < 2) {
    const int j = lane;
    const size_t base = (size_t)j * VOCAB;
    float lin = fm_b[j]
              + fm_w[base + uid]
              + fm_w[base + NUM_USERS + hr]
              + fm_w[base + NUM_USERS + 24 + gd]
              + fm_w[base + NUM_USERS + 24 + 2 + ag]
              + fm_w[base + NUM_USERS + 24 + 2 + NUM_AGES + at];
    fm[b * 2 + j] = lin + 0.5f * so;
  }
}

// ---------------- WMMA helpers ----------------
__device__ __forceinline__ v16bf make_a_frag(v8bf lo, v8bf hi) {
  v16bf a;
#pragma unroll
  for (int i = 0; i < 8; ++i) { a[i] = lo[i]; a[i + 8] = hi[i]; }
  return a;
}

// ---------------- visual GEMM: [B,2048] f32  x  visu_w[64,2048] bf16 -> x0[:,320:384] ----------------
// block = 128 rows, 8 waves; wave w: 16 rows x 64 cols (4 wmma accumulators)
__global__ __launch_bounds__(256) void visual_gemm_kernel(
    const float* __restrict__ X, const bf16_t* __restrict__ W,
    const float* __restrict__ bias, bf16_t* __restrict__ Y) {
  const int w    = threadIdx.x >> 5;
  const int lane = threadIdx.x & 31;
  const int half = lane >> 4;     // 0: K 0-15 ; 1: K 16-31
  const int lr   = lane & 15;
  const int m_base = blockIdx.x * 128 + w * 16;

  v8f acc[4] = {{}, {}, {}, {}};
  const float* xrow = X + (size_t)(m_base + lr) * VISUAL_DIM;
  const bf16_t* wrow[4];
#pragma unroll
  for (int t = 0; t < 4; ++t)
    wrow[t] = W + (size_t)(t * 16 + lr) * VISUAL_DIM + half * 16;

  for (int kb = 0; kb < VISUAL_DIM; kb += 32) {
    // A fragment (fp32 -> bf16 in-register)
    const float* xp = xrow + kb + half * 16;
    v8f flo = *(const v8f*)(xp);
    v8f fhi = *(const v8f*)(xp + 8);
    if (kb + 32 < VISUAL_DIM) __builtin_prefetch(xp + 32, 0, 0);
    v16bf a = make_a_frag(__builtin_convertvector(flo, v8bf),
                          __builtin_convertvector(fhi, v8bf));
#pragma unroll
    for (int t = 0; t < 4; ++t) {
      v16bf bfr = *(const v16bf*)(wrow[t] + kb);
      acc[t] = __builtin_amdgcn_wmma_f32_16x16x32_bf16(false, a, false, bfr,
                                                       (short)0, acc[t], false, false);
    }
  }

#pragma unroll
  for (int t = 0; t < 4; ++t) {
    const int n = t * 16 + lr;
    const float bi = bias[n];
#pragma unroll
    for (int i = 0; i < 8; ++i) {
      const int m = m_base + i + half * 8;
      Y[(size_t)m * X0_LD + 5 * FACE_K + n] = (bf16_t)(acc[t][i] + bi);
    }
  }
}

// ---------------- generic bf16 GEMM + bias + optional relu ----------------
// Y[M, ldy] (cols 0..N-1) = relu(X[M,K] * W[N,K]^T + bias)
// grid: (M/128, N/64); 8 waves, wave w: 16 rows x 64 cols (4 accumulators)
__global__ __launch_bounds__(256) void gemm_bf16_kernel(
    const bf16_t* __restrict__ X, const bf16_t* __restrict__ W,
    const float* __restrict__ bias, bf16_t* __restrict__ Y,
    int K, int ldy, int do_relu) {
  const int w    = threadIdx.x >> 5;
  const int lane = threadIdx.x & 31;
  const int half = lane >> 4;
  const int lr   = lane & 15;
  const int m_base = blockIdx.x * 128 + w * 16;
  const int n_base = blockIdx.y * 64;

  v8f acc[4] = {{}, {}, {}, {}};
  const bf16_t* xrow = X + (size_t)(m_base + lr) * K + half * 16;
  const bf16_t* wrow[4];
#pragma unroll
  for (int t = 0; t < 4; ++t)
    wrow[t] = W + (size_t)(n_base + t * 16 + lr) * K + half * 16;

  for (int kb = 0; kb < K; kb += 32) {
    v8bf alo = *(const v8bf*)(xrow + kb);
    v8bf ahi = *(const v8bf*)(xrow + kb + 8);
    v16bf a  = make_a_frag(alo, ahi);
#pragma unroll
    for (int t = 0; t < 4; ++t) {
      v16bf bfr = *(const v16bf*)(wrow[t] + kb);
      acc[t] = __builtin_amdgcn_wmma_f32_16x16x32_bf16(false, a, false, bfr,
                                                       (short)0, acc[t], false, false);
    }
  }

#pragma unroll
  for (int t = 0; t < 4; ++t) {
    const int n = n_base + t * 16 + lr;
    const float bi = bias[n];
#pragma unroll
    for (int i = 0; i < 8; ++i) {
      const int m = m_base + i + half * 8;
      float v = acc[t][i] + bi;
      if (do_relu) v = fmaxf(v, 0.0f);
      Y[(size_t)m * ldy + n] = (bf16_t)v;
    }
  }
}

// ---------------- final: layer3 (N=2) + fm + softmax ----------------
__global__ __launch_bounds__(256) void final_kernel(
    const bf16_t* __restrict__ x2, const float* __restrict__ w3,
    const float* __restrict__ b3, const float* __restrict__ fm,
    float* __restrict__ out) {
  const int wv   = threadIdx.x >> 5;
  const int lane = threadIdx.x & 31;
  const int b    = blockIdx.x * 8 + wv;
  if (b >= BATCH) return;

  const bf16_t* xr = x2 + (size_t)b * HIDDEN;
  float d0 = 0.0f, d1 = 0.0f;
#pragma unroll
  for (int k = lane; k < HIDDEN; k += 32) {
    const float xv = (float)xr[k];
    d0 += xv * w3[k];
    d1 += xv * w3[HIDDEN + k];
  }
#pragma unroll
  for (int off = 16; off > 0; off >>= 1) {
    d0 += __shfl_xor(d0, off, 32);
    d1 += __shfl_xor(d1, off, 32);
  }
  if (lane == 0) {
    const float l0 = fm[b * 2 + 0] + d0 + b3[0];
    const float l1 = fm[b * 2 + 1] + d1 + b3[1];
    const float mx = fmaxf(l0, l1);
    const float e0 = expf(l0 - mx);
    const float e1 = expf(l1 - mx);
    const float inv = 1.0f / (e0 + e1);
    out[b * 2 + 0] = e0 * inv;
    out[b * 2 + 1] = e1 * inv;
  }
}

// ---------------- launch ----------------
extern "C" void kernel_launch(void* const* d_in, const int* in_sizes, int n_in,
                              void* d_out, int out_size, void* d_ws, size_t ws_size,
                              hipStream_t stream) {
  (void)in_sizes; (void)n_in; (void)out_size; (void)ws_size;

  const int*   user_id    = (const int*)d_in[0];
  const int*   hour       = (const int*)d_in[1];
  const float* visual     = (const float*)d_in[2];
  const float* scale      = (const float*)d_in[3];
  const int*   gender     = (const int*)d_in[4];
  const int*   age        = (const int*)d_in[5];
  const int*   attribute  = (const int*)d_in[6];
  const float* user_emb   = (const float*)d_in[7];
  const float* hour_emb   = (const float*)d_in[8];
  const float* gender_emb = (const float*)d_in[9];
  const float* age_emb    = (const float*)d_in[10];
  const float* attr_emb   = (const float*)d_in[11];
  const float* visu_w     = (const float*)d_in[12];
  const float* visu_b     = (const float*)d_in[13];
  const float* fm_w       = (const float*)d_in[14];
  const float* fm_b       = (const float*)d_in[15];
  const float* w1         = (const float*)d_in[16];
  const float* b1         = (const float*)d_in[17];
  const float* w2         = (const float*)d_in[18];
  const float* b2         = (const float*)d_in[19];
  const float* w3         = (const float*)d_in[20];
  const float* b3         = (const float*)d_in[21];

  char* ws = (char*)d_ws;
  float*  fm  = (float*)(ws + OFF_FM);
  bf16_t* x0  = (bf16_t*)(ws + OFF_X0);
  bf16_t* x1  = (bf16_t*)(ws + OFF_X1);
  bf16_t* x2  = (bf16_t*)(ws + OFF_X2);
  bf16_t* wvb = (bf16_t*)(ws + OFF_WV);
  bf16_t* w1b = (bf16_t*)(ws + OFF_W1);
  bf16_t* w2b = (bf16_t*)(ws + OFF_W2);

  // weight conversions (tiny)
  cvt_f32_bf16<<<256, 256, 0, stream>>>(visu_w, wvb, FACE_K * VISUAL_DIM);
  cvt_f32_bf16<<<256, 256, 0, stream>>>(w1, w1b, HIDDEN * X0_LD);
  cvt_f32_bf16<<<256, 256, 0, stream>>>(w2, w2b, HIDDEN * HIDDEN);

  // embeddings + FM linear + second order -> x0[:,0:320], fm
  embed_fm_kernel<<<BATCH / 8, 256, 0, stream>>>(
      user_id, hour, gender, age, attribute, scale,
      user_emb, hour_emb, gender_emb, age_emb, attr_emb,
      fm_w, fm_b, x0, fm);

  // visual projection -> x0[:,320:384]
  visual_gemm_kernel<<<BATCH / 128, 256, 0, stream>>>(visual, wvb, visu_b, x0);

  // layer1: x1 = relu(x0 @ w1^T + b1)   [K=384, N=512]
  gemm_bf16_kernel<<<dim3(BATCH / 128, HIDDEN / 64), 256, 0, stream>>>(
      x0, w1b, b1, x1, X0_LD, HIDDEN, 1);

  // layer2: x2 = relu(x1 @ w2^T + b2)   [K=512, N=512]
  gemm_bf16_kernel<<<dim3(BATCH / 128, HIDDEN / 64), 256, 0, stream>>>(
      x1, w2b, b2, x2, HIDDEN, HIDDEN, 1);

  // layer3 + fm + softmax
  final_kernel<<<BATCH / 8, 256, 0, stream>>>(x2, w3, b3, fm, (float*)d_out);
}
